// magnn_43602507989304
// MI455X (gfx1250) — matrix operand order repaired and checked by reading, not compiled, not run on hardware
//
#include <hip/hip_runtime.h>
#include <hip/hip_bf16.h>
#include <math.h>

// ---------------------------------------------------------------------------
// MAGNN-style multi-scale GNN forward for MI455X (gfx1250, wave32, WMMA).
// Heavy GEMMs (adjacency build, mixprop diffusion, channel projection) run on
// v_wmma_f32_16x16x32_f16. All WMMA operands are stored in f16 (converted at
// production, not per K-chunk), laid out so fragments are direct b128 loads:
// adjacency operands pre-transposed (K = v contiguous), activations v-minor
// (B, C, L, NP). All inner loops are branch-free via zero-padded buffers.
// f32 is kept where reference numerics need it: alpha-mix residual, G
// accumulation, adjacency output, top-k, and the head MLPs.
// ---------------------------------------------------------------------------

typedef _Float16 v16h __attribute__((ext_vector_type(16)));
typedef _Float16 v8h  __attribute__((ext_vector_type(8)));
typedef float    v8f  __attribute__((ext_vector_type(8)));

#define WMMA_F16(A, B, C) \
  __builtin_amdgcn_wmma_f32_16x16x32_f16(false, (A), false, (B), (short)0, (C), false, false)

#define CAT16(lo, hi) \
  __builtin_shufflevector((lo), (hi), 0, 1, 2, 3, 4, 5, 6, 7, 8, 9, 10, 11, 12, 13, 14, 15)

static constexpr int   Bb   = 4;     // batch
static constexpr int   Nn   = 600;   // nodes (valid)
static constexpr int   NP   = 608;   // nodes padded: 19*32 -> guard-free K and M
static constexpr int   Ss   = 168;   // input sequence length
static constexpr int   Cc   = 32;    // conv channels
static constexpr int   Csk  = 16;    // skip channels
static constexpr int   Ce   = 128;   // end channels
static constexpr int   Od   = 12;    // output dim
static constexpr int   Dd   = 40;    // embedding dim (valid)
static constexpr int   KP   = 64;    // embedding dim padded: 2*32 chunks
static constexpr int   TOPK = 20;
static constexpr float ALPHA = 0.05f;
static constexpr float TA    = 3.0f;

// --------------------------- zero fill ---------------------------
__global__ void k_zero(float* __restrict__ p, long n) {
  long i = (long)blockIdx.x * blockDim.x + threadIdx.x;
  if (i < n) p[i] = 0.f;
}

// ------------------- stage 0: 1x1 start conv (layout B,Cc,Ss,NP) ------------
__global__ void k_start(const float* __restrict__ x, const float* __restrict__ Ws,
                        const float* __restrict__ bs, float* __restrict__ h0) {
  int idx = blockIdx.x * blockDim.x + threadIdx.x;          // over B*Ss*Nn, n fastest
  if (idx >= Bb * Ss * Nn) return;
  int n = idx % Nn, l = (idx / Nn) % Ss, b = idx / (Nn * Ss);
  float xv = x[((size_t)b * Nn + n) * Ss + l];
  for (int o = 0; o < Cc; ++o)
    h0[(((size_t)b * Cc + o) * Ss + l) * NP + n] = xv * Ws[o] + bs[o];
}

// ------- stage 1: strided temporal conv, n-fastest; dual f32 + f16 store ----
__global__ void k_tconv(const float* __restrict__ in, const float* __restrict__ W,
                        const float* __restrict__ bias, float* __restrict__ out,
                        _Float16* __restrict__ out16, int Lin, int Lout, int K,
                        int stride) {
  int n = blockIdx.x * blockDim.x + threadIdx.x;
  if (n >= Nn) return;
  int p = blockIdx.y;                 // over Cc*Lout
  int o = p / Lout, t = p % Lout;
  int b = blockIdx.z;
  float acc = bias[o];
  int base = t * stride;
  for (int c = 0; c < Cc; ++c) {
    const float* ip = &in[(((size_t)b * Cc + c) * Lin + base) * NP + n];
    const float* wp = &W[((size_t)o * Cc + c) * K];
    for (int k = 0; k < K; ++k) acc += ip[(size_t)k * NP] * wp[k];
  }
  size_t oidx = (((size_t)b * Cc + o) * Lout + t) * NP + n;
  out[oidx]   = acc;
  out16[oidx] = (_Float16)acc;
}

// ---------------------- stage 2: time-pooled features ----------------------
__global__ void k_feat(const float* __restrict__ s, float* __restrict__ feat, int Lw) {
  int idx = blockIdx.x * blockDim.x + threadIdx.x;          // over B*Cc*Nn, n fastest
  if (idx >= Bb * Cc * Nn) return;
  int n = idx % Nn, c = (idx / Nn) % Cc, b = idx / (Nn * Cc);
  const float* p = &s[(((size_t)b * Cc + c) * Lw) * NP + n];
  float acc = 0.f;
  for (int t = 0; t < Lw; ++t) acc += p[(size_t)t * NP];
  feat[((size_t)b * Nn + n) * Cc + c] = acc / (float)Lw;
}

// ----------------------- stage 3: node embeddings n1/n2 (f16) ---------------
__global__ void k_nodes(const float* __restrict__ feat,
                        const float* __restrict__ W1, const float* __restrict__ b1,
                        const float* __restrict__ W2, const float* __restrict__ b2,
                        const float* __restrict__ e1, const float* __restrict__ e2,
                        float sfac, _Float16* __restrict__ n1,
                        _Float16* __restrict__ n2) {
  int idx = blockIdx.x * blockDim.x + threadIdx.x;          // over B*N*Dd
  if (idx >= Bb * Nn * Dd) return;
  int d = idx % Dd, n = (idx / Dd) % Nn, b = idx / (Dd * Nn);
  const float* f = &feat[((size_t)b * Nn + n) * Cc];
  float a1 = b1[d] + e1[n * Dd + d];
  float a2 = b2[d] + e2[n * Dd + d];
  for (int c = 0; c < Cc; ++c) {
    float fv = f[c];
    a1 += fv * W1[c * Dd + d];
    a2 += fv * W2[c * Dd + d];
  }
  size_t o = ((size_t)b * NP + n) * KP + d;
  n1[o] = (_Float16)tanhf(TA * sfac * a1);
  n2[o] = (_Float16)tanhf(TA * sfac * a2);
}

// ------------- stage 4: adjacency  relu(tanh(3*(n1 n2^T - n2 n1^T))) --------
// f16 operands: A frag = 2x b128 (v8h pair), B frag = 1x v16h (2x b128).
__global__ void __launch_bounds__(32) k_adj(const _Float16* __restrict__ n1,
                                            const _Float16* __restrict__ n2,
                                            float* __restrict__ adj) {
  int lane = threadIdx.x & 31, half = lane >> 4, lan = lane & 15;
  int b = blockIdx.z;
  int r0 = blockIdx.x * 16, c0 = blockIdx.y * 16;
  const _Float16* p1 = n1 + (size_t)b * NP * KP;
  const _Float16* p2 = n2 + (size_t)b * NP * KP;
  int arow = r0 + lan;                    // < NP, always valid
  int bcol = c0 + lan;
  const v8h* r1 = reinterpret_cast<const v8h*>(p1 + (size_t)arow * KP);
  const v8h* r2 = reinterpret_cast<const v8h*>(p2 + (size_t)arow * KP);
  const _Float16* c1 = p2 + (size_t)bcol * KP;
  const _Float16* c2 = p1 + (size_t)bcol * KP;
  v8f acc1 = {0.f,0.f,0.f,0.f,0.f,0.f,0.f,0.f};
  v8f acc2 = {0.f,0.f,0.f,0.f,0.f,0.f,0.f,0.f};
#pragma unroll
  for (int k0 = 0; k0 < KP; k0 += 32) {
    int ia = (k0 + half * 16) >> 3;       // v8h index; +8 halves = +1
    int ib = k0 + half * 16;              // contiguous 16 halves
    v16h a1  = CAT16(r1[ia], r1[ia + 1]);
    v16h a2  = CAT16(r2[ia], r2[ia + 1]);
    v16h b1f = *reinterpret_cast<const v16h*>(c1 + ib);
    v16h b2f = *reinterpret_cast<const v16h*>(c2 + ib);
    acc1 = WMMA_F16(a1, b1f, acc1);
    acc2 = WMMA_F16(a2, b2f, acc2);
  }
#pragma unroll
  for (int r = 0; r < 8; ++r) {
    int rr = r0 + r + 8 * half, cc = c0 + lan;
    if (rr < Nn && cc < Nn) {
      float v = tanhf(TA * (acc1[r] - acc2[r]));
      adj[((size_t)b * Nn + rr) * Nn + cc] = v > 0.f ? v : 0.f;
    }
  }
}

// ------ stage 5: per-row top-k mask (bit-exact 20th-largest via bisection) --
__global__ void k_topk(float* __restrict__ adj) {
  int v = blockIdx.x, b = blockIdx.y, tid = threadIdx.x;    // 256 threads
  float* row = adj + ((size_t)b * Nn + v) * Nn;
  float vals[3];
  int nv = 0;
  for (int i = tid; i < Nn; i += 256) vals[nv++] = row[i];
  __shared__ int cnt;
  unsigned lo = 0u, hi = 0x3F800000u;     // values in [0,1)
  while (hi - lo > 1u) {
    unsigned mid = (lo + hi) >> 1;
    float t = __uint_as_float(mid);
    if (tid == 0) cnt = 0;
    __syncthreads();
    int c = 0;
    for (int i = 0; i < nv; ++i) c += (vals[i] >= t) ? 1 : 0;
    atomicAdd(&cnt, c);
    __syncthreads();
    int total = cnt;
    __syncthreads();
    if (total >= TOPK) lo = mid; else hi = mid;
  }
  float kth = __uint_as_float(lo);
  for (int i = 0; i < nv; ++i) {
    int idx = tid + i * 256;
    row[idx] = (vals[i] >= kth) ? vals[i] : 0.f;
  }
}

// ------ stage 6a: rinv[v] = 1/(1+rowsum), cinv[v] = 1/(1+colsum) ------------
__global__ void k_sums(const float* __restrict__ adj, float* __restrict__ rinv,
                       float* __restrict__ cinv) {
  int v = blockIdx.x, b = blockIdx.y, tid = threadIdx.x;    // 256 threads
  const float* base = adj + (size_t)b * Nn * Nn;
  __shared__ float rs[256], cs[256];
  float r = 0.f, c = 0.f;
  for (int w = tid; w < Nn; w += 256) {
    r += base[(size_t)v * Nn + w];
    c += base[(size_t)w * Nn + v];
  }
  rs[tid] = r; cs[tid] = c;
  __syncthreads();
  for (int s2 = 128; s2 > 0; s2 >>= 1) {
    if (tid < s2) { rs[tid] += rs[tid + s2]; cs[tid] += cs[tid + s2]; }
    __syncthreads();
  }
  if (tid == 0) {
    rinv[(size_t)b * NP + v] = 1.f / (1.f + rs[0]);
    cinv[(size_t)b * NP + v] = 1.f / (1.f + cs[0]);
  }
}

// ------ stage 6b: pre-transposed normalized adjacencies, f16 (WMMA A layout)
// At0[w][v] = (adj[v][w]+d)/rowsum[v]   (dir 0: rownorm(adj+I), transposed)
// At1[w][v] = (adj[w][v]+d)/colsum[v]   (dir 1: rownorm(adj^T+I), transposed)
__global__ void k_trans(const float* __restrict__ adj, const float* __restrict__ rinv,
                        const float* __restrict__ cinv, _Float16* __restrict__ At0,
                        _Float16* __restrict__ At1) {
  int w = blockIdx.x, b = blockIdx.y, tid = threadIdx.x;    // 256 threads
  const float* base = adj + (size_t)b * Nn * Nn;
  for (int v = tid; v < Nn; v += 256) {
    float d = (v == w) ? 1.f : 0.f;
    At0[((size_t)b * NP + w) * NP + v] =
        (_Float16)((base[(size_t)v * Nn + w] + d) * rinv[(size_t)b * NP + v]);
    At1[((size_t)b * NP + w) * NP + v] =
        (_Float16)((base[(size_t)w * Nn + v] + d) * cinv[(size_t)b * NP + v]);
  }
}

// ------ stage 7: diffusion  Hd[c,t,w] = a*X + (1-a) * sum_v At[w,v] Hs[c,t,v]
// f16 operands and f16 output; f32 residual X. One wave: 16 rows (w) x 32 cols
// (two (c,t) tiles sharing the A fragment). All memory ops are b128.
__global__ void __launch_bounds__(32) k_mixprop(const _Float16* __restrict__ At,
                                                const _Float16* __restrict__ Hs,
                                                const float* __restrict__ X,
                                                _Float16* __restrict__ Hd, int Lw) {
  int lane = threadIdx.x & 31, half = lane >> 4, lan = lane & 15;
  int b = blockIdx.z;
  int w0 = blockIdx.x * 16;
  int j0 = blockIdx.y * 32;
  int jA = j0 + lan, jB = j0 + 16 + lan;        // two column tiles, < Cc*Lw
  int cA = jA / Lw, tA = jA % Lw;
  int cB = jB / Lw, tB = jB % Lw;
  const v8h* arow = reinterpret_cast<const v8h*>(
      At + ((size_t)b * NP + (w0 + lan)) * NP);
  size_t plA = (((size_t)b * Cc + cA) * Lw + tA) * NP;
  size_t plB = (((size_t)b * Cc + cB) * Lw + tB) * NP;
  const _Float16* hA = Hs + plA;
  const _Float16* hB = Hs + plB;
  v8f acc0 = {0.f,0.f,0.f,0.f,0.f,0.f,0.f,0.f};
  v8f acc1 = {0.f,0.f,0.f,0.f,0.f,0.f,0.f,0.f};
  for (int k0 = 0; k0 < NP; k0 += 32) {
    if (k0 + 32 < NP) {
      __builtin_prefetch(&arow[(k0 + 32) >> 3], 0, 1);
      __builtin_prefetch(hA + k0 + 32, 0, 1);
    }
    int ia = (k0 + half * 16) >> 3;             // v8h index; +8 halves = +1
    int ib = k0 + half * 16;                    // contiguous 16 halves
    v16h af  = CAT16(arow[ia], arow[ia + 1]);
    v16h bf0 = *reinterpret_cast<const v16h*>(hA + ib);
    v16h bf1 = *reinterpret_cast<const v16h*>(hB + ib);
    acc0 = WMMA_F16(af, bf0, acc0);
    acc1 = WMMA_F16(af, bf1, acc1);
  }
  // D rows w0+8*half .. +7 are contiguous in the v-minor layout
  int wb = w0 + 8 * half;
  {
    const float4* x4 = reinterpret_cast<const float4*>(X + plA + wb);
    float4 xa = x4[0], xb = x4[1];
    v8h o;
    o[0] = (_Float16)(ALPHA * xa.x + (1.f - ALPHA) * acc0[0]);
    o[1] = (_Float16)(ALPHA * xa.y + (1.f - ALPHA) * acc0[1]);
    o[2] = (_Float16)(ALPHA * xa.z + (1.f - ALPHA) * acc0[2]);
    o[3] = (_Float16)(ALPHA * xa.w + (1.f - ALPHA) * acc0[3]);
    o[4] = (_Float16)(ALPHA * xb.x + (1.f - ALPHA) * acc0[4]);
    o[5] = (_Float16)(ALPHA * xb.y + (1.f - ALPHA) * acc0[5]);
    o[6] = (_Float16)(ALPHA * xb.z + (1.f - ALPHA) * acc0[6]);
    o[7] = (_Float16)(ALPHA * xb.w + (1.f - ALPHA) * acc0[7]);
    *reinterpret_cast<v8h*>(Hd + plA + wb) = o;
  }
  {
    const float4* x4 = reinterpret_cast<const float4*>(X + plB + wb);
    float4 xa = x4[0], xb = x4[1];
    v8h o;
    o[0] = (_Float16)(ALPHA * xa.x + (1.f - ALPHA) * acc1[0]);
    o[1] = (_Float16)(ALPHA * xa.y + (1.f - ALPHA) * acc1[1]);
    o[2] = (_Float16)(ALPHA * xa.z + (1.f - ALPHA) * acc1[2]);
    o[3] = (_Float16)(ALPHA * xa.w + (1.f - ALPHA) * acc1[3]);
    o[4] = (_Float16)(ALPHA * xb.x + (1.f - ALPHA) * acc1[4]);
    o[5] = (_Float16)(ALPHA * xb.y + (1.f - ALPHA) * acc1[5]);
    o[6] = (_Float16)(ALPHA * xb.z + (1.f - ALPHA) * acc1[6]);
    o[7] = (_Float16)(ALPHA * xb.w + (1.f - ALPHA) * acc1[7]);
    *reinterpret_cast<v8h*>(Hd + plB + wb) = o;
  }
}

// ------ stage 8: channel projection  G[o,nl] (+)= sum_{k<96} ho[k,nl] W[k,o]+b
// K = 96 aligns with the three f16 sources {X16, H1, H2}. One wave computes
// both 16-wide o tiles sharing the A fragment; M = flat (t,v) over Lw*NP.
__global__ void __launch_bounds__(32) k_proj(const _Float16* __restrict__ X16,
                                             const _Float16* __restrict__ H1,
                                             const _Float16* __restrict__ H2,
                                             const float* __restrict__ W,
                                             const float* __restrict__ bias,
                                             float* __restrict__ G, int Lw,
                                             int accumulate) {
  int lane = threadIdx.x & 31, half = lane >> 4, lan = lane & 15;
  int b = blockIdx.z;
  int NL = Lw * NP;
  int nl0 = blockIdx.x * 16;
  int o0 = lan, o1 = 16 + lan;
  const _Float16* srcs[3] = { X16 + (size_t)b * Cc * NL,
                              H1 + (size_t)b * Cc * NL,
                              H2 + (size_t)b * Cc * NL };
  int nlA = nl0 + lan;
  v8f acc0 = {0.f,0.f,0.f,0.f,0.f,0.f,0.f,0.f};
  v8f acc1 = {0.f,0.f,0.f,0.f,0.f,0.f,0.f,0.f};
#pragma unroll
  for (int s2 = 0; s2 < 3; ++s2) {
    const _Float16* base = srcs[s2];
    v16h af, bf0, bf1;
#pragma unroll
    for (int e = 0; e < 8; ++e)
      af[e] = base[(size_t)(half * 16 + e) * NL + nlA];
#pragma unroll
    for (int e = 0; e < 8; ++e)
      af[8 + e] = base[(size_t)(half * 16 + 8 + e) * NL + nlA];
#pragma unroll
    for (int e = 0; e < 16; ++e) {
      int ch = s2 * 32 + half * 16 + e;
      bf0[e] = (_Float16)W[(size_t)ch * Cc + o0];
      bf1[e] = (_Float16)W[(size_t)ch * Cc + o1];
    }
    acc0 = WMMA_F16(af, bf0, acc0);
    acc1 = WMMA_F16(af, bf1, acc1);
  }
  int nlb = nl0 + 8 * half;                     // rows contiguous across r
  {
    float b0 = bias[o0];
    float4 oa, ob;
    oa.x = acc0[0] + b0; oa.y = acc0[1] + b0; oa.z = acc0[2] + b0; oa.w = acc0[3] + b0;
    ob.x = acc0[4] + b0; ob.y = acc0[5] + b0; ob.z = acc0[6] + b0; ob.w = acc0[7] + b0;
    float4* g4 = reinterpret_cast<float4*>(G + ((size_t)b * Cc + o0) * NL + nlb);
    if (accumulate) {
      float4 p0 = g4[0], p1 = g4[1];
      oa.x += p0.x; oa.y += p0.y; oa.z += p0.z; oa.w += p0.w;
      ob.x += p1.x; ob.y += p1.y; ob.z += p1.z; ob.w += p1.w;
    }
    g4[0] = oa; g4[1] = ob;
  }
  {
    float b1 = bias[o1];
    float4 oa, ob;
    oa.x = acc1[0] + b1; oa.y = acc1[1] + b1; oa.z = acc1[2] + b1; oa.w = acc1[3] + b1;
    ob.x = acc1[4] + b1; ob.y = acc1[5] + b1; ob.z = acc1[6] + b1; ob.w = acc1[7] + b1;
    float4* g4 = reinterpret_cast<float4*>(G + ((size_t)b * Cc + o1) * NL + nlb);
    if (accumulate) {
      float4 p0 = g4[0], p1 = g4[1];
      oa.x += p0.x; oa.y += p0.y; oa.z += p0.z; oa.w += p0.w;
      ob.x += p1.x; ob.y += p1.y; ob.z += p1.z; ob.w += p1.w;
    }
    g4[0] = oa; g4[1] = ob;
  }
}

// ------------------- stage 9: skip head  out[b,o,n] --------------------------
__global__ void k_skip(const float* __restrict__ G, const float* __restrict__ svW,
                       const float* __restrict__ svb, float* __restrict__ outslot,
                       int Lw) {
  int n = blockIdx.x * blockDim.x + threadIdx.x;
  if (n >= Nn) return;
  int o = blockIdx.y, b = blockIdx.z;
  float acc = svb[o];
  for (int c = 0; c < Cc; ++c) {
    const float* g = &G[(((size_t)b * Cc + c) * Lw) * NP + n];
    const float* w = &svW[((size_t)o * Cc + c) * Lw];
    for (int t = 0; t < Lw; ++t) acc += g[(size_t)t * NP] * w[t];
  }
  outslot[((size_t)b * Csk + o) * Nn + n] = acc;
}

__global__ void k_out0(const float* __restrict__ x, const float* __restrict__ s0W,
                       const float* __restrict__ s0b, float* __restrict__ outslot) {
  int n = blockIdx.x * blockDim.x + threadIdx.x;
  if (n >= Nn) return;
  int o = blockIdx.y, b = blockIdx.z;
  float acc = s0b[o];
  const float* xp = &x[((size_t)b * Nn + n) * Ss];
  const float* w  = &s0W[(size_t)o * Ss];
  for (int t = 0; t < Ss; ++t) acc += xp[t] * w[t];
  outslot[((size_t)b * Csk + o) * Nn + n] = acc;
}

// ------------------- stage 10: SE gate over the 4 scale heads ----------------
__global__ void k_se(const float* __restrict__ outs, const float* __restrict__ gfW1,
                     const float* __restrict__ gfW2, float* __restrict__ se) {
  int b = blockIdx.x, tid = threadIdx.x;                    // 64 threads
  __shared__ float sm[64];
  __shared__ float z1[4];
  int slot = tid / Csk, o = tid % Csk;
  const float* p = &outs[(((size_t)slot * Bb + b) * Csk + o) * Nn];
  float a = 0.f;
  for (int n = 0; n < Nn; ++n) a += p[n];
  sm[tid] = a / (float)Nn;
  __syncthreads();
  if (tid < 4) {
    float z = 0.f;
    for (int j2 = 0; j2 < 64; ++j2) z += sm[j2] * gfW1[j2 * 4 + tid];
    z1[tid] = z > 0.f ? z : 0.f;
  }
  __syncthreads();
  if (tid < 4) {
    float z = 0.f;
    for (int q = 0; q < 4; ++q) z += z1[q] * gfW2[q * 4 + tid];
    se[b * 4 + tid] = 1.f / (1.f + expf(-z));
  }
}

// ------------------- stage 11: fuse + end MLP -> y ---------------------------
__global__ void k_final(const float* __restrict__ outs, const float* __restrict__ se,
                        const float* __restrict__ e1W, const float* __restrict__ e1b,
                        const float* __restrict__ e2W, const float* __restrict__ e2b,
                        float* __restrict__ y) {
  int n = blockIdx.x, b = blockIdx.y, tid = threadIdx.x;    // 128 threads
  __shared__ float fs[Csk];
  __shared__ float h[Ce];
  if (tid < Csk) {
    float a = 0.f;
    for (int i2 = 0; i2 < 4; ++i2)
      a += outs[(((size_t)i2 * Bb + b) * Csk + tid) * Nn + n] * se[b * 4 + i2];
    a *= 0.25f;
    fs[tid] = a > 0.f ? a : 0.f;
  }
  __syncthreads();
  {
    float a = e1b[tid];
    for (int cs2 = 0; cs2 < Csk; ++cs2) a += e1W[tid * Csk + cs2] * fs[cs2];
    h[tid] = a > 0.f ? a : 0.f;
  }
  __syncthreads();
  if (tid < Od) {
    float a = e2b[tid];
    for (int ce = 0; ce < Ce; ++ce) a += e2W[tid * Ce + ce] * h[ce];
    y[((size_t)b * Od + tid) * Nn + n] = a;
  }
}

// ---------------------------------------------------------------------------
extern "C" void kernel_launch(void* const* d_in, const int* in_sizes, int n_in,
                              void* d_out, int out_size, void* d_ws, size_t ws_size,
                              hipStream_t stream) {
  (void)in_sizes; (void)n_in; (void)out_size; (void)ws_size;
  const float* x     = (const float*)d_in[0];
  const float* Ws    = (const float*)d_in[1];
  const float* bs    = (const float*)d_in[2];
  const float* scWp[3] = { (const float*)d_in[3], (const float*)d_in[4], (const float*)d_in[5] };
  const float* scb   = (const float*)d_in[6];
  const float* emb1  = (const float*)d_in[7];
  const float* emb2  = (const float*)d_in[8];
  const float* lin1W = (const float*)d_in[9];
  const float* lin1b = (const float*)d_in[10];
  const float* lin2W = (const float*)d_in[11];
  const float* lin2b = (const float*)d_in[12];
  const float* gW    = (const float*)d_in[13];
  const float* gb    = (const float*)d_in[14];
  const float* svWp[3] = { (const float*)d_in[15], (const float*)d_in[16], (const float*)d_in[17] };
  const float* svb   = (const float*)d_in[18];
  const float* s0W   = (const float*)d_in[19];
  const float* s0b   = (const float*)d_in[20];
  const float* gfW1  = (const float*)d_in[21];
  const float* gfW2  = (const float*)d_in[22];
  const float* e1W   = (const float*)d_in[23];
  const float* e1b   = (const float*)d_in[24];
  const float* e2W   = (const float*)d_in[25];
  const float* e2b   = (const float*)d_in[26];

  float* yout = (float*)d_out;                       // (B,12,N,1)
  float* adjs = yout + (size_t)Bb * Od * Nn;         // (3,B,N,N)

  const int Lws[3] = {162, 79, 39};
  const int Kk[3]  = {7, 6, 3};
  const int Str[3] = {1, 2, 2};
  const float sfacs[3] = {1.0f, 0.8f, 0.6f};

  const size_t SZH0 = (size_t)Bb * Cc * Ss * NP;      // (B,Cc,Ss,NP) f32
  const size_t SZ0  = (size_t)Bb * Cc * Lws[0] * NP;  // element counts
  const size_t SZ1  = (size_t)Bb * Cc * Lws[1] * NP;
  const size_t SZ2  = (size_t)Bb * Cc * Lws[2] * NP;
  const size_t SZN  = (size_t)Bb * NP * KP;           // n1/n2 (f16)
  const size_t SZA  = (size_t)Bb * NP * NP;           // At0/At1 (f16)

  // ---- f32 region ----
  float* ws   = (float*)d_ws;
  float* h0   = ws;                                   // SZH0; H1(f16) aliases later
  float* s[3];
  s[0] = h0 + SZH0;
  s[1] = s[0] + SZ0;
  s[2] = s[1] + SZ1;
  float* G    = s[2] + SZ2;                           // SZ0
  float* feat = G + SZ0;                              // B*Nn*Cc
  float* rinv = feat + (size_t)Bb * Nn * Cc;          // B*NP
  float* cinv = rinv + (size_t)Bb * NP;
  float* outs = cinv + (size_t)Bb * NP;               // 4*B*Csk*Nn
  float* se   = outs + (size_t)4 * Bb * Csk * Nn;     // B*4
  // ---- f16 region (element counts above, 2 bytes each) ----
  _Float16* s16[3];
  s16[0] = (_Float16*)(se + Bb * 4);
  s16[1] = s16[0] + SZ0;
  s16[2] = s16[1] + SZ1;
  _Float16* H2h  = s16[2] + SZ2;                      // SZ0 halves
  _Float16* n1h  = H2h + SZ0;
  _Float16* n2h  = n1h + SZN;
  _Float16* At0h = n2h + SZN;
  _Float16* At1h = At0h + SZA;
  _Float16* H1h  = (_Float16*)h0;                     // alias: h0 dead after tconvs

  // zero the padded buffers whose pads feed the guard-free WMMA loops:
  // f32 scales, f16 scales+H2+n1+n2+At0+At1 (contiguous halves; 0x0000 = +0.0h)
  {
    long zn = (long)(SZ0 + SZ1 + SZ2);
    k_zero<<<dim3((unsigned)((zn + 255) / 256)), dim3(256), 0, stream>>>(s[0], zn);
    long zh = (long)((2 * (SZ0 + SZ1 + SZ2) + 2 * SZ0 + 2 * SZN + 2 * SZA) / 2) / 2;
    // halves total = (SZ0+SZ1+SZ2) + SZ0 + 2*SZN + 2*SZA ; as floats = /2
    zh = (long)(((SZ0 + SZ1 + SZ2) + SZ0 + 2 * SZN + 2 * SZA) / 2);
    k_zero<<<dim3((unsigned)((zh + 255) / 256)), dim3(256), 0, stream>>>(
        (float*)s16[0], zh);
  }

  // stage 0: start conv
  {
    int tot = Bb * Ss * Nn;
    k_start<<<dim3((tot + 255) / 256), dim3(256), 0, stream>>>(x, Ws, bs, h0);
  }
  // stage 1: three strided temporal convs (n-fastest, coalesced, dual store)
  {
    const float* tin = h0;
    int Lin = Ss;
    for (int i = 0; i < 3; ++i) {
      k_tconv<<<dim3((Nn + 255) / 256, Cc * Lws[i], Bb), dim3(256), 0, stream>>>(
          tin, scWp[i], scb + i * Cc, s[i], s16[i], Lin, Lws[i], Kk[i], Str[i]);
      tin = s[i];
      Lin = Lws[i];
    }
  }

  for (int i = 0; i < 3; ++i) {
    int Lw = Lws[i];
    {
      int tot = Bb * Cc * Nn;
      k_feat<<<dim3((tot + 255) / 256), dim3(256), 0, stream>>>(s[i], feat, Lw);
    }
    {
      int tot = Bb * Nn * Dd;
      k_nodes<<<dim3((tot + 255) / 256), dim3(256), 0, stream>>>(
          feat, lin1W + (size_t)i * Cc * Dd, lin1b + i * Dd,
          lin2W + (size_t)i * Cc * Dd, lin2b + i * Dd,
          emb1, emb2, sfacs[i], n1h, n2h);
    }
    // adjacency (WMMA) -> top-k mask (in d_out) -> transposed normalized A's
    float* adj_i = adjs + (size_t)i * Bb * Nn * Nn;
    k_adj  <<<dim3(NP / 16, NP / 16, Bb), dim3(32), 0, stream>>>(n1h, n2h, adj_i);
    k_topk <<<dim3(Nn, Bb), dim3(256), 0, stream>>>(adj_i);
    k_sums <<<dim3(Nn, Bb), dim3(256), 0, stream>>>(adj_i, rinv, cinv);
    k_trans<<<dim3(Nn, Bb), dim3(256), 0, stream>>>(adj_i, rinv, cinv, At0h, At1h);

    // mixprop diffusion (WMMA) + channel projection (WMMA), both directions
    int NJt = (Cc * Lw) / 32;                 // exact for all scales
    int Mt  = (Lw * NP) / 16;                 // exact: NP multiple of 16
    for (int d2 = 0; d2 < 2; ++d2) {
      const _Float16* Ad = d2 ? At1h : At0h;
      k_mixprop<<<dim3(NP / 16, NJt, Bb), dim3(32), 0, stream>>>(Ad, s16[i], s[i], H1h, Lw);
      k_mixprop<<<dim3(NP / 16, NJt, Bb), dim3(32), 0, stream>>>(Ad, H1h, s[i], H2h, Lw);
      k_proj<<<dim3(Mt, 1, Bb), dim3(32), 0, stream>>>(
          s16[i], H1h, H2h,
          gW + ((size_t)i * 2 + d2) * 96 * Cc,
          gb + ((size_t)i * 2 + d2) * Cc,
          G, Lw, d2);
    }
    k_skip<<<dim3((Nn + 255) / 256, Csk, Bb), dim3(256), 0, stream>>>(
        G, svWp[i], svb + i * Csk, outs + (size_t)(i + 1) * Bb * Csk * Nn, Lw);
  }
  k_out0<<<dim3((Nn + 255) / 256, Csk, Bb), dim3(256), 0, stream>>>(x, s0W, s0b, outs);
  k_se<<<dim3(Bb), dim3(64), 0, stream>>>(outs, gfW1, gfW2, se);
  k_final<<<dim3(Nn, Bb), dim3(128), 0, stream>>>(outs, se, e1W, e1b, e2W, e2b, yout);
}
